// BobaTransformerBlock_28561532518578
// MI455X (gfx1250) — matrix operand order repaired and checked by reading, not compiled, not run on hardware
//
#include <hip/hip_runtime.h>
#include <hip/hip_bf16.h>

// ---------------------------------------------------------------------------
// BobaTransformerBlock on MI455X (gfx1250, wave32, WMMA).
// All GEMMs via v_wmma_f32_16x16x32_bf16; fp32 accumulation + residuals.
// M=32 row tiles, B-fragment shared across two row sub-tiles.
// k/v are written TRANSPOSED from the qkv GEMM epilogue so the kv-outer-product
// GEMM (K=8192) uses only contiguous b128 loads; that GEMM is split-K x4 for
// occupancy with a deterministic fixed-order reduction.
// ---------------------------------------------------------------------------

typedef __attribute__((ext_vector_type(16))) __bf16 v16bf;
typedef __attribute__((ext_vector_type(8)))  __bf16 v8bf;
typedef __attribute__((ext_vector_type(8)))  float  v8f;

__device__ __forceinline__ __bf16 f2bf(float f) {
  unsigned u = __builtin_bit_cast(unsigned, f);
  u += 0x7FFFu + ((u >> 16) & 1u);          // round-to-nearest-even
  unsigned short h = (unsigned short)(u >> 16);
  return __builtin_bit_cast(__bf16, h);
}

__device__ __forceinline__ v16bf cat8(v8bf lo, v8bf hi) {
  return __builtin_shufflevector(lo, hi, 0,1,2,3,4,5,6,7,8,9,10,11,12,13,14,15);
}

// A fragment (16xK tile, 16-bit): lane&15 = row M; lanes 0-15 hold K 0..15,
// lanes 16-31 hold K 16..31.  src row-major, ld elements per row.
__device__ __forceinline__ v16bf load_a(const __bf16* src, int ld, int k0, int lane) {
  int row = lane & 15;
  int ks  = (lane >> 4) << 4;               // 0 or 16
  return *(const v16bf*)(src + (size_t)row * ld + k0 + ks);
}

// B fragment (Kx16 tile): lane&15 = column N; lanes 0-15 hold K 0..15,
// lanes 16-31 hold K 16..31.  Operand stored with K contiguous per output
// column so this is one contiguous 32B load per lane.
__device__ __forceinline__ v16bf load_b(const __bf16* w, int ld, int col0, int k0, int lane) {
  int col = col0 + (lane & 15);
  int ks  = (lane >> 4) << 4;               // 0 or 16
  return *(const v16bf*)(w + (size_t)col * ld + k0 + ks);
}

__device__ __forceinline__ v8f wmma_bf16(v16bf a, v16bf b, v8f c) {
  return __builtin_amdgcn_wmma_f32_16x16x32_bf16(false, a, false, b, (short)0, c,
                                                 false, false);
}

// Two row sub-tiles (rows 0-15 and 16-31 of a 32-row A tile) sharing each B frag.
template <int K>
__device__ __forceinline__ void gemm_2row(const __bf16* aT, int lda,
                                          const __bf16* w, int ldw, int col0,
                                          int lane, v8f& acc0, v8f& acc1) {
#pragma unroll
  for (int kk = 0; kk < K; kk += 32) {
    __builtin_prefetch(w + (size_t)(col0 + (lane & 15)) * ldw + kk + 64, 0, 1);
    v16bf b  = load_b(w, ldw, col0, kk, lane);
    v16bf a0 = load_a(aT, lda, kk, lane);
    v16bf a1 = load_a(aT + (size_t)16 * lda, lda, kk, lane);
    acc0 = wmma_bf16(a0, b, acc0);
    acc1 = wmma_bf16(a1, b, acc1);
  }
}

// Stage a 32x256 fp32 tile scaled by gamma into LDS as bf16 (256 threads).
__device__ __forceinline__ void stage32(const float* __restrict__ x,
                                        const float* __restrict__ gamma,
                                        __bf16* lds, int tid) {
  int row = tid >> 3;              // 32 rows, 8 threads/row
  int col = (tid & 7) << 5;        // 32 cols each
  const float* p = x + (size_t)row * 256 + col;
  __bf16* q = lds + row * 256 + col;
#pragma unroll
  for (int i = 0; i < 32; ++i) q[i] = f2bf(p[i] * gamma[col + i]);
}

// --------------------------- kernel 0: fp32 -> bf16 weights -----------------
__global__ void k_cvt(const float* __restrict__ src, __bf16* __restrict__ dst, int n) {
  int i = blockIdx.x * blockDim.x + threadIdx.x;
  if (i < n) dst[i] = f2bf(src[i]);
}

// ------------------- kernel 1: qkv = (x*g1) @ Wqkv^T ------------------------
// grid 2048 x 256.  32-row tile; 8 waves x 12 col-tiles of 1536.
// q  -> qb[m][512]           (row-major, token-major)
// k  -> kT[head][d][n]       (transposed: contiguous-n 16B store per lane)
// v  -> vT[head][e][n]       (transposed)
__global__ __launch_bounds__(256) void k_qkv(const float* __restrict__ x,
                                             const float* __restrict__ gamma1,
                                             const __bf16* __restrict__ wqkv,
                                             __bf16* __restrict__ qb,
                                             __bf16* __restrict__ kT,
                                             __bf16* __restrict__ vT) {
  __shared__ __align__(32) __bf16 a_tile[32 * 256];
  int m0  = blockIdx.x * 32;
  int b   = m0 >> 13;
  int n0  = m0 & 8191;
  int tid = threadIdx.x;
  stage32(x + (size_t)m0 * 256, gamma1, a_tile, tid);
  __syncthreads();
  int wave = tid >> 5, lane = tid & 31;
  for (int t = 0; t < 12; ++t) {
    int o0 = (wave * 12 + t) * 16;
    v8f acc0 = {}, acc1 = {};
    gemm_2row<256>(a_tile, 256, wqkv, 256, o0, lane, acc0, acc1);
    int lc = lane & 15, rbase = (lane >> 4) << 3;
    if (o0 < 512) {                          // q: row-major store
      int col = o0 + lc;
#pragma unroll
      for (int r = 0; r < 8; ++r) {
        qb[(size_t)(m0 + rbase + r) * 512 + col]      = f2bf(acc0[r]);
        qb[(size_t)(m0 + 16 + rbase + r) * 512 + col] = f2bf(acc1[r]);
      }
    } else {                                 // k/v: transposed 16B stores
      int local = o0 - 512;
      __bf16* dstT = (local < 512) ? kT : vT;
      int loc  = local & 511;
      int h    = loc >> 6;
      int dcol = (loc & 63) + lc;
      __bf16* base = dstT + ((size_t)(b * 8 + h)) * 64 * 8192
                          + (size_t)dcol * 8192 + n0 + rbase;
      v8bf s0, s1;
#pragma unroll
      for (int r = 0; r < 8; ++r) { s0[r] = f2bf(acc0[r]); s1[r] = f2bf(acc1[r]); }
      *(v8bf*)(base)      = s0;
      *(v8bf*)(base + 16) = s1;
    }
  }
}

// ------------- kernel 2a: partial kv[b,h] = k^T @ v  (split-K x4) -----------
// grid 1024 x 32: one wave per (head, e-tile, split); each split reduces 2048
// tokens.  B frag reused across 4 d-tiles.  Partials stored fp32, e-major.
__global__ __launch_bounds__(32) void k_kv(const __bf16* __restrict__ kT,
                                           const __bf16* __restrict__ vT,
                                           float* __restrict__ kvp) {
  int head  = blockIdx.x >> 4;              // b*8+h in [0,64)
  int e0    = ((blockIdx.x >> 2) & 3) << 4;
  int split = blockIdx.x & 3;
  int lane  = threadIdx.x;
  int nbeg  = split * 2048;
  const __bf16* kA = kT + (size_t)head * 64 * 8192;
  const __bf16* vB = vT + (size_t)head * 64 * 8192;
  v8f acc[4] = {{}, {}, {}, {}};
  for (int n = nbeg; n < nbeg + 2048; n += 32) {
    __builtin_prefetch(vB + (size_t)(e0 + (lane & 15)) * 8192 + n + 64, 0, 1);
    v16bf b = load_b(vB, 8192, e0, n, lane);
#pragma unroll
    for (int dt = 0; dt < 4; ++dt) {
      v16bf a = load_a(kA + (size_t)dt * 16 * 8192, 8192, n, lane);
      acc[dt] = wmma_bf16(a, b, acc[dt]);
    }
  }
  int rbase = (lane >> 4) << 3;
  int ecol  = e0 + (lane & 15);
  float* base = kvp + ((size_t)split * 64 + head) * 4096 + (size_t)ecol * 64;
#pragma unroll
  for (int dt = 0; dt < 4; ++dt)
    *(v8f*)(base + dt * 16 + rbase) = acc[dt];
}

// ------------- kernel 2b: kvT = bf16( sum over 4 splits ) -------------------
// Fixed-order sum => deterministic.  grid 1024 x 256.
__global__ __launch_bounds__(256) void k_kvred(const float* __restrict__ kvp,
                                               __bf16* __restrict__ kvT) {
  int i = blockIdx.x * blockDim.x + threadIdx.x;   // [0, 64*4096)
  float s = kvp[i];
  s += kvp[(size_t)1 * 64 * 4096 + i];
  s += kvp[(size_t)2 * 64 * 4096 + i];
  s += kvp[(size_t)3 * 64 * 4096 + i];
  kvT[i] = f2bf(s);
}

// ------------- kernel 3: x2 = x + ((q @ kv) * s) @ Wout^T + b_out -----------
// grid 2048 x 256.  Phase 1: t(32x512) bf16 in LDS.  Phase 2: project + residual.
__global__ __launch_bounds__(256) void k_attn(const float* __restrict__ x,
                                              const __bf16* __restrict__ qb,
                                              const __bf16* __restrict__ kvT,
                                              const __bf16* __restrict__ wout,
                                              const float* __restrict__ b_out,
                                              float* __restrict__ x2) {
  __shared__ __align__(32) __bf16 t_tile[32 * 512];
  int m0  = blockIdx.x * 32;
  int b   = m0 >> 13;
  int tid = threadIdx.x, wave = tid >> 5, lane = tid & 31;
  const float s = (1.0f / 8.0f) / 8192.0f;  // SCALE / N
  for (int t = 0; t < 4; ++t) {
    int c0 = (wave * 4 + t) * 16;           // column in [0,512)
    int h  = c0 >> 6;
    int e0 = c0 & 63;
    const __bf16* kvh = kvT + (size_t)(b * 8 + h) * 4096;
    const __bf16* qA0 = qb + (size_t)m0 * 512 + h * 64;
    const __bf16* qA1 = qb + (size_t)(m0 + 16) * 512 + h * 64;
    v8f acc0 = {}, acc1 = {};
#pragma unroll
    for (int kk = 0; kk < 64; kk += 32) {
      v16bf bm = load_b(kvh, 64, e0, kk, lane);
      v16bf a0 = load_a(qA0, 512, kk, lane);
      v16bf a1 = load_a(qA1, 512, kk, lane);
      acc0 = wmma_bf16(a0, bm, acc0);
      acc1 = wmma_bf16(a1, bm, acc1);
    }
    int col   = c0 + (lane & 15);
    int rbase = (lane >> 4) << 3;
#pragma unroll
    for (int r = 0; r < 8; ++r) {
      t_tile[(rbase + r) * 512 + col]      = f2bf(acc0[r] * s);
      t_tile[(16 + rbase + r) * 512 + col] = f2bf(acc1[r] * s);
    }
  }
  __syncthreads();
  for (int t = 0; t < 2; ++t) {
    int o0 = (wave * 2 + t) * 16;
    v8f acc0 = {}, acc1 = {};
    gemm_2row<512>(t_tile, 512, wout, 512, o0, lane, acc0, acc1);
    int col   = o0 + (lane & 15);
    int rbase = (lane >> 4) << 3;
    float bo  = b_out[col];
#pragma unroll
    for (int r = 0; r < 8; ++r) {
      size_t i0 = (size_t)(m0 + rbase + r) * 256 + col;
      size_t i1 = (size_t)(m0 + 16 + rbase + r) * 256 + col;
      x2[i0] = x[i0] + acc0[r] + bo;
      x2[i1] = x[i1] + acc1[r] + bo;
    }
  }
}

// --------- kernel 4: out = x2 + gelu((x2*g2) @ W1^T + b1) @ W2^T + b2 -------
// grid 2048 x 256.  Fully fused MLP.  LDS: 16KB a-tile + 64KB m-tile.
__global__ __launch_bounds__(256) void k_mlp(const float* __restrict__ x2,
                                             const float* __restrict__ gamma2,
                                             const __bf16* __restrict__ w1,
                                             const float* __restrict__ b1,
                                             const __bf16* __restrict__ w2,
                                             const float* __restrict__ b2,
                                             float* __restrict__ out) {
  __shared__ __align__(32) __bf16 a_tile[32 * 256];
  __shared__ __align__(32) __bf16 m_tile[32 * 1024];
  int m0  = blockIdx.x * 32;
  int tid = threadIdx.x, wave = tid >> 5, lane = tid & 31;
  stage32(x2 + (size_t)m0 * 256, gamma2, a_tile, tid);
  __syncthreads();
  for (int t = 0; t < 8; ++t) {
    int c0 = (wave * 8 + t) * 16;           // column in [0,1024)
    v8f acc0 = {}, acc1 = {};
    gemm_2row<256>(a_tile, 256, w1, 256, c0, lane, acc0, acc1);
    int col   = c0 + (lane & 15);
    int rbase = (lane >> 4) << 3;
    float bb  = b1[col];
#pragma unroll
    for (int r = 0; r < 8; ++r) {
      float m0v = acc0[r] + bb;
      float m1v = acc1[r] + bb;
      float g0  = 0.5f * m0v + m0v * m0v * m0v * (0.1972f + 0.0012f * m0v);
      float g1  = 0.5f * m1v + m1v * m1v * m1v * (0.1972f + 0.0012f * m1v);
      m_tile[(rbase + r) * 1024 + col]      = f2bf(g0);
      m_tile[(16 + rbase + r) * 1024 + col] = f2bf(g1);
    }
  }
  __syncthreads();
  for (int t = 0; t < 2; ++t) {
    int o0 = (wave * 2 + t) * 16;
    v8f acc0 = {}, acc1 = {};
    gemm_2row<1024>(m_tile, 1024, w2, 1024, o0, lane, acc0, acc1);
    int col   = o0 + (lane & 15);
    int rbase = (lane >> 4) << 3;
    float bb  = b2[col];
#pragma unroll
    for (int r = 0; r < 8; ++r) {
      size_t i0 = (size_t)(m0 + rbase + r) * 256 + col;
      size_t i1 = (size_t)(m0 + 16 + rbase + r) * 256 + col;
      out[i0] = x2[i0] + acc0[r] + bb;
      out[i1] = x2[i1] + acc1[r] + bb;
    }
  }
}

// ---------------------------------------------------------------------------
extern "C" void kernel_launch(void* const* d_in, const int* in_sizes, int n_in,
                              void* d_out, int out_size, void* d_ws, size_t ws_size,
                              hipStream_t stream) {
  const float* x    = (const float*)d_in[0];
  const float* g1   = (const float*)d_in[1];
  const float* wqkv = (const float*)d_in[2];
  const float* wout = (const float*)d_in[3];
  const float* bout = (const float*)d_in[4];
  const float* g2   = (const float*)d_in[5];
  const float* w1   = (const float*)d_in[6];
  const float* b1   = (const float*)d_in[7];
  const float* w2   = (const float*)d_in[8];
  const float* b2   = (const float*)d_in[9];
  float* out = (float*)d_out;

  char* ws = (char*)d_ws;
  size_t off = 0;
  auto alloc = [&](size_t bytes) {
    void* p = ws + off;
    off = (off + bytes + 255) & ~(size_t)255;
    return p;
  };
  __bf16* wqkv_b = (__bf16*)alloc((size_t)1536 * 256 * 2);
  __bf16* wout_b = (__bf16*)alloc((size_t)256 * 512 * 2);
  __bf16* w1_b   = (__bf16*)alloc((size_t)1024 * 256 * 2);
  __bf16* w2_b   = (__bf16*)alloc((size_t)256 * 1024 * 2);
  __bf16* qb     = (__bf16*)alloc((size_t)65536 * 512 * 2);
  __bf16* kT     = (__bf16*)alloc((size_t)64 * 64 * 8192 * 2);
  __bf16* vT     = (__bf16*)alloc((size_t)64 * 64 * 8192 * 2);
  __bf16* kvT    = (__bf16*)alloc((size_t)64 * 64 * 64 * 2);
  float*  kvp    = (float*) alloc((size_t)4 * 64 * 4096 * 4);
  float*  x2f    = (float*) alloc((size_t)65536 * 256 * 4);
  (void)ws_size; (void)in_sizes; (void)n_in; (void)out_size;

  k_cvt<<<(393216 + 255) / 256, 256, 0, stream>>>(wqkv, wqkv_b, 393216);
  k_cvt<<<(131072 + 255) / 256, 256, 0, stream>>>(wout, wout_b, 131072);
  k_cvt<<<(262144 + 255) / 256, 256, 0, stream>>>(w1, w1_b, 262144);
  k_cvt<<<(262144 + 255) / 256, 256, 0, stream>>>(w2, w2_b, 262144);

  k_qkv  <<<2048, 256, 0, stream>>>(x, g1, wqkv_b, qb, kT, vT);
  k_kv   <<<1024, 32,  0, stream>>>(kT, vT, kvp);
  k_kvred<<<1024, 256, 0, stream>>>(kvp, kvT);
  k_attn <<<2048, 256, 0, stream>>>(x, qb, kvT, wout_b, bout, x2f);
  k_mlp  <<<2048, 256, 0, stream>>>(x2f, g2, w1_b, b1, w2_b, b2, out);
}
